// imageCaptionModel_12833362281086
// MI455X (gfx1250) — compile-verified, matrix operand order checked by reading, not run on hardware
//
#include <hip/hip_runtime.h>
#include <hip/hip_bf16.h>
#include <math.h>

#define BATCH   128
#define SEQ     32
#define HID     512
#define EMB     512
#define CNN_F   2048
#define VOCAB   10000

typedef __attribute__((ext_vector_type(16))) __bf16 v16bf;
typedef __attribute__((ext_vector_type(8)))  float  v8f;
typedef unsigned short u16;

// ---------------- helpers ----------------

__device__ __forceinline__ u16 f32_to_bf16(float f) {
  union { float f; unsigned int u; } in; in.f = f;
  unsigned int u = in.u;
  unsigned int r = u + 0x7FFFu + ((u >> 16) & 1u);   // round-to-nearest-even
  return (u16)(r >> 16);
}

__device__ __forceinline__ float sigmoidf_(float x) {
  return 1.0f / (1.0f + __expf(-x));
}

// ---- fragment-swizzled storage ----
// Every WMMA operand lives in memory as [tile][lane(32)][16 bf16], i.e. the
// exact per-lane register image of ISA 7.12.2 (16-bit 16x32 A / 32x16 B):
//   lane = (m_or_n % 16) + 16*hi,  hi = ((k%16) >= 8)
//   slot = 8*((k%32) >= 16) + (k % 8)
// so a fragment load is two global_load_b128 per lane.

__device__ __forceinline__ int swz_off(int tile, int lane, int slot) {
  return ((tile << 5) + lane) * 16 + slot;
}

// A-operand index for element (m, k); tile order: mt-major, kt-minor.
__device__ __forceinline__ int a_swz(int m, int k, int ktiles) {
  const int kr = k & 31, j = kr & 15;
  const int tile = (m >> 4) * ktiles + (k >> 5);
  return swz_off(tile, (m & 15) + ((j >> 3) << 4), ((kr >> 4) << 3) + (j & 7));
}

__device__ __forceinline__ v16bf load_frag(const u16* __restrict__ base, int tile,
                                           int lane) {
  union { v16bf v; uint4 q[2]; } f;
  const uint4* p = (const uint4*)(base + (size_t)swz_off(tile, lane, 0));
  f.q[0] = p[0];
  f.q[1] = p[1];
  return f.v;
}

__device__ __forceinline__ v8f wmma_bf16(v16bf a, v16bf b, v8f c) {
  return __builtin_amdgcn_wmma_f32_16x16x32_bf16(false, a, false, b, (short)0, c,
                                                 false, false);
}

// ---------------- convert + swizzle kernels ----------------

// A-operand (activations) swizzle: src f32 [M, K] row-major
__global__ void k_swz_a(const float* __restrict__ src, u16* __restrict__ dst,
                        int M, int K) {
  int i = blockIdx.x * blockDim.x + threadIdx.x;
  if (i >= M * K) return;
  int m = i / K, k = i - m * K;
  dst[a_swz(m, k, K >> 5)] = f32_to_bf16(src[i]);
}

// B-operand (weights) swizzle: src f32 [K, N] row-major; tile order kt-major,
// nt-minor (consecutive nt contiguous for the multi-N inner loop).
__global__ void k_swz_b(const float* __restrict__ src, u16* __restrict__ dst,
                        int K, int N) {
  int i = blockIdx.x * blockDim.x + threadIdx.x;
  if (i >= K * N) return;
  int k = i / N, n = i - k * N;
  const int kr = k & 31, j = kr & 15;
  const int tile = (k >> 5) * (N >> 4) + (n >> 4);
  dst[swz_off(tile, (n & 15) + ((j >> 3) << 4), ((kr >> 4) << 3) + (j & 7))] =
      f32_to_bf16(src[i]);
}

// xemb (A-swizzled per step): xemb[t] holds [B=128, E=512] fragments
__global__ void k_gather(const float* __restrict__ emb, const int* __restrict__ tok,
                         u16* __restrict__ xemb) {
  int i = blockIdx.x * blockDim.x + threadIdx.x;
  if (i >= SEQ * BATCH * EMB) return;
  int e = i % EMB;
  int b = (i / EMB) % BATCH;
  int t = i / (EMB * BATCH);
  int token = tok[b * SEQ + t];
  xemb[(size_t)t * BATCH * EMB + a_swz(b, e, EMB >> 5)] =
      f32_to_bf16(emb[(long)token * EMB + e]);
}

__global__ void k_copy_final(const float* __restrict__ h, float* __restrict__ dst) {
  int i = blockIdx.x * blockDim.x + threadIdx.x;
  if (i < 2 * BATCH * HID) dst[i] = h[i];
}

// ---------------- WMMA kernels ----------------
// 256 threads = 8 waves per block; grids cover tiles exactly (EXEC all-1s).

#define KT_H (HID >> 5)     // 16 k-tiles per 512-wide segment
#define NT_H (HID >> 4)     // 32 n-tiles of hid

// h0 = tanh(cnn @ in_W + in_b), replicated to both layers (f32 linear + bf16 swz)
__global__ void k_h0(const u16* __restrict__ cnn_swz, const u16* __restrict__ inW_swz,
                     const float* __restrict__ in_b,
                     float* __restrict__ h_f32, u16* __restrict__ h_swz) {
  const int wave = threadIdx.x >> 5, lane = threadIdx.x & 31;
  const int tile = blockIdx.x * 8 + wave;          // 8 x 32 = 256 tiles
  const int mt = tile >> 5, nt = tile & 31;
  v8f acc = {};
  for (int kt = 0; kt < (CNN_F >> 5); ++kt) {
    v16bf a = load_frag(cnn_swz, mt * (CNN_F >> 5) + kt, lane);
    v16bf b = load_frag(inW_swz, kt * NT_H + nt, lane);
    acc = wmma_bf16(a, b, acc);
  }
  const int col = nt * 16 + (lane & 15);
  const int rbase = mt * 16 + ((lane >> 4) << 3);
  const float bias = in_b[col];
#pragma unroll
  for (int v = 0; v < 8; ++v) {
    const int row = rbase + v;
    const int idx = row * HID + col;
    float h = tanhf(acc[v] + bias);
    u16 hb = f32_to_bf16(h);
    const int sw = a_swz(row, col, KT_H);
    h_f32[idx] = h;  h_f32[BATCH * HID + idx] = h;
    h_swz[sw]  = hb; h_swz[BATCH * HID + sw]  = hb;
  }
}

// u = sigmoid([x,h]@Wu+bu), r = sigmoid([x,h]@Wr+br); stores u (f32) and swz bf16(r*h)
__global__ void k_gates_ur(const u16* __restrict__ x_swz, const u16* __restrict__ h_swz,
                           const u16* __restrict__ Wu, const u16* __restrict__ Wr,
                           const float* __restrict__ bu, const float* __restrict__ br,
                           const float* __restrict__ h_f32,
                           float* __restrict__ u_out, u16* __restrict__ rh_swz) {
  const int wave = threadIdx.x >> 5, lane = threadIdx.x & 31;
  const int tile = blockIdx.x * 8 + wave;
  const int mt = tile >> 5, nt = tile & 31;
  v8f au = {}, ar = {};
#pragma unroll
  for (int s = 0; s < 2; ++s) {
    const u16* A = (s == 0) ? x_swz : h_swz;
    for (int kt = 0; kt < KT_H; ++kt) {
      v16bf a  = load_frag(A, mt * KT_H + kt, lane);
      v16bf b0 = load_frag(Wu, (s * KT_H + kt) * NT_H + nt, lane);
      v16bf b1 = load_frag(Wr, (s * KT_H + kt) * NT_H + nt, lane);
      au = wmma_bf16(a, b0, au);
      ar = wmma_bf16(a, b1, ar);
    }
  }
  const int col = nt * 16 + (lane & 15);
  const int rbase = mt * 16 + ((lane >> 4) << 3);
  const float biu = bu[col], bir = br[col];
#pragma unroll
  for (int v = 0; v < 8; ++v) {
    const int row = rbase + v;
    const int idx = row * HID + col;
    float u = sigmoidf_(au[v] + biu);
    float r = sigmoidf_(ar[v] + bir);
    u_out[idx] = u;
    rh_swz[a_swz(row, col, KT_H)] = f32_to_bf16(r * h_f32[idx]);
  }
}

// c = tanh([x, r*h]@Wc + bc); h' = u*h + (1-u)*c  (in-place h update)
__global__ void k_gates_c(const u16* __restrict__ x_swz, const u16* __restrict__ rh_swz,
                          const u16* __restrict__ Wc, const float* __restrict__ bc,
                          const float* __restrict__ u_in,
                          float* __restrict__ h_f32, u16* __restrict__ h_swz,
                          u16* __restrict__ hseq_swz /* may be null */) {
  const int wave = threadIdx.x >> 5, lane = threadIdx.x & 31;
  const int tile = blockIdx.x * 8 + wave;
  const int mt = tile >> 5, nt = tile & 31;
  v8f acc = {};
#pragma unroll
  for (int s = 0; s < 2; ++s) {
    const u16* A = (s == 0) ? x_swz : rh_swz;
    for (int kt = 0; kt < KT_H; ++kt) {
      v16bf a = load_frag(A, mt * KT_H + kt, lane);
      v16bf b = load_frag(Wc, (s * KT_H + kt) * NT_H + nt, lane);
      acc = wmma_bf16(a, b, acc);
    }
  }
  const int col = nt * 16 + (lane & 15);
  const int rbase = mt * 16 + ((lane >> 4) << 3);
  const float bias = bc[col];
#pragma unroll
  for (int v = 0; v < 8; ++v) {
    const int row = rbase + v;
    const int idx = row * HID + col;
    float c = tanhf(acc[v] + bias);
    float u = u_in[idx];
    float h = u * h_f32[idx] + (1.0f - u) * c;
    u16 hb = f32_to_bf16(h);
    const int sw = a_swz(row, col, KT_H);
    h_f32[idx] = h;
    h_swz[sw]  = hb;
    if (hseq_swz) hseq_swz[sw] = hb;   // per-step base applied by caller
  }
}

// logits: [T*B, H] @ [H, V] + out_b, written in [B, T, V] order.
// 5 N-tiles per wave (625 = 5 * 125): per K-step 2 A-loads + 10 B-loads / 5 WMMA.
#define LOGIT_NT 5
#define LOGIT_NG (VOCAB / 16 / LOGIT_NT)   // 125 n-groups
__global__ void k_logits(const u16* __restrict__ hseq_swz, const u16* __restrict__ outW_swz,
                         const float* __restrict__ out_b, float* __restrict__ out) {
  const int wave = threadIdx.x >> 5, lane = threadIdx.x & 31;
  const int task = blockIdx.x * 8 + wave;        // 256 * 125 = 32000 tasks
  const int mt = task / LOGIT_NG;                // 0..255
  const int ng = task - mt * LOGIT_NG;
  const int nt0 = ng * LOGIT_NT;
  v8f acc[LOGIT_NT] = {};
  for (int kt = 0; kt < KT_H; ++kt) {
    v16bf a = load_frag(hseq_swz, mt * KT_H + kt, lane);
    __builtin_prefetch(outW_swz + (size_t)swz_off((kt + 1) * (VOCAB >> 4) + nt0, lane, 0), 0, 0);
#pragma unroll
    for (int i = 0; i < LOGIT_NT; ++i) {
      v16bf b = load_frag(outW_swz, kt * (VOCAB >> 4) + nt0 + i, lane);
      acc[i] = wmma_bf16(a, b, acc[i]);
    }
  }
  const int rbase = mt * 16 + ((lane >> 4) << 3);
#pragma unroll
  for (int i = 0; i < LOGIT_NT; ++i) {
    const int col = (nt0 + i) * 16 + (lane & 15);
    const float bias = out_b[col];
#pragma unroll
    for (int v = 0; v < 8; ++v) {
      const int m = rbase + v;                   // m = t*BATCH + b
      const int t = m >> 7, b = m & 127;
      out[((long)(b * SEQ + t)) * VOCAB + col] = acc[i][v] + bias;
    }
  }
}

// ---------------- host ----------------

extern "C" void kernel_launch(void* const* d_in, const int* in_sizes, int n_in,
                              void* d_out, int out_size, void* d_ws, size_t ws_size,
                              hipStream_t stream) {
  const float* cnn   = (const float*)d_in[0];
  const int*   tok   = (const int*)  d_in[1];
  const float* emb   = (const float*)d_in[2];
  const float* in_W  = (const float*)d_in[3];
  const float* in_b  = (const float*)d_in[4];
  const float* Wu0   = (const float*)d_in[5];
  const float* bu0   = (const float*)d_in[6];
  const float* Wr0   = (const float*)d_in[7];
  const float* br0   = (const float*)d_in[8];
  const float* Wc0   = (const float*)d_in[9];
  const float* bc0   = (const float*)d_in[10];
  const float* Wu1   = (const float*)d_in[11];
  const float* bu1   = (const float*)d_in[12];
  const float* Wr1   = (const float*)d_in[13];
  const float* br1   = (const float*)d_in[14];
  const float* Wc1   = (const float*)d_in[15];
  const float* bc1   = (const float*)d_in[16];
  const float* out_W = (const float*)d_in[17];
  const float* out_b = (const float*)d_in[18];

  char* ws = (char*)d_ws;
  size_t off = 0;
  auto carve = [&](size_t bytes) -> void* {
    void* p = ws + off;
    off += (bytes + 255) & ~(size_t)255;
    return p;
  };
  u16* cnn_swz  = (u16*)carve((size_t)BATCH * CNN_F * 2);
  u16* inW_swz  = (u16*)carve((size_t)CNN_F * HID * 2);
  u16* Wu0_swz  = (u16*)carve((size_t)(EMB + HID) * HID * 2);
  u16* Wr0_swz  = (u16*)carve((size_t)(EMB + HID) * HID * 2);
  u16* Wc0_swz  = (u16*)carve((size_t)(EMB + HID) * HID * 2);
  u16* Wu1_swz  = (u16*)carve((size_t)(HID + HID) * HID * 2);
  u16* Wr1_swz  = (u16*)carve((size_t)(HID + HID) * HID * 2);
  u16* Wc1_swz  = (u16*)carve((size_t)(HID + HID) * HID * 2);
  u16* outW_swz = (u16*)carve((size_t)HID * VOCAB * 2);
  u16* xemb     = (u16*)carve((size_t)SEQ * BATCH * EMB * 2);
  u16* hseq_swz = (u16*)carve((size_t)SEQ * BATCH * HID * 2);
  float* h_f32  = (float*)carve((size_t)2 * BATCH * HID * 4);
  u16* h_swz    = (u16*)carve((size_t)2 * BATCH * HID * 2);
  float* u_buf  = (float*)carve((size_t)BATCH * HID * 4);
  u16* rh_swz   = (u16*)carve((size_t)BATCH * HID * 2);
  (void)ws_size; (void)n_in; (void)in_sizes; (void)out_size;

  auto swzA = [&](const float* s, u16* d, int M, int K) {
    k_swz_a<<<(M * K + 255) / 256, 256, 0, stream>>>(s, d, M, K);
  };
  auto swzB = [&](const float* s, u16* d, int K, int N) {
    k_swz_b<<<(K * N + 255) / 256, 256, 0, stream>>>(s, d, K, N);
  };
  swzA(cnn, cnn_swz, BATCH, CNN_F);
  swzB(in_W,  inW_swz,  CNN_F, HID);
  swzB(Wu0,   Wu0_swz,  EMB + HID, HID);
  swzB(Wr0,   Wr0_swz,  EMB + HID, HID);
  swzB(Wc0,   Wc0_swz,  EMB + HID, HID);
  swzB(Wu1,   Wu1_swz,  HID + HID, HID);
  swzB(Wr1,   Wr1_swz,  HID + HID, HID);
  swzB(Wc1,   Wc1_swz,  HID + HID, HID);
  swzB(out_W, outW_swz, HID, VOCAB);
  k_gather<<<(SEQ * BATCH * EMB + 255) / 256, 256, 0, stream>>>(emb, tok, xemb);

  // h0 -> both layers
  k_h0<<<32, 256, 0, stream>>>(cnn_swz, inW_swz, in_b, h_f32, h_swz);

  // recurrence: 32 steps x 2 layers
  for (int t = 0; t < SEQ; ++t) {
    const u16* x0 = xemb + (size_t)t * BATCH * EMB;
    // layer 0
    k_gates_ur<<<32, 256, 0, stream>>>(x0, h_swz, Wu0_swz, Wr0_swz, bu0, br0,
                                       h_f32, u_buf, rh_swz);
    k_gates_c<<<32, 256, 0, stream>>>(x0, rh_swz, Wc0_swz, bc0, u_buf,
                                      h_f32, h_swz, (u16*)nullptr);
    // layer 1 (input = layer-0 hidden just produced)
    k_gates_ur<<<32, 256, 0, stream>>>(h_swz, h_swz + BATCH * HID, Wu1_swz, Wr1_swz,
                                       bu1, br1, h_f32 + BATCH * HID, u_buf, rh_swz);
    k_gates_c<<<32, 256, 0, stream>>>(h_swz, rh_swz, Wc1_swz, bc1, u_buf,
                                      h_f32 + BATCH * HID, h_swz + BATCH * HID,
                                      hseq_swz + (size_t)t * BATCH * HID);
  }

  // dominant GEMM: 32000 wave-tasks / 8 waves per block
  k_logits<<<4000, 256, 0, stream>>>(hseq_swz, outW_swz, out_b, (float*)d_out);

  // final_state [2, B, H]
  k_copy_final<<<(2 * BATCH * HID + 255) / 256, 256, 0, stream>>>(
      h_f32, (float*)d_out + (size_t)BATCH * SEQ * VOCAB);
}
